// PosLearnedSimulator_43009802502737
// MI455X (gfx1250) — matrix operand order repaired and checked by reading, not compiled, not run on hardware
//
#include <hip/hip_runtime.h>

// ---------------------------------------------------------------------------
// MI455X (gfx1250) fused GNN: wave32 WMMA f16->f32 GEMM tiles, LDS-staged
// activations, fused gather / scatter-add epilogues.
// 32-row M-tiles per block: each weight (B) fragment feeds 2 WMMAs.
// ---------------------------------------------------------------------------

typedef __attribute__((ext_vector_type(16))) _Float16 v16h;
typedef __attribute__((ext_vector_type(8)))  _Float16 v8h;
typedef __attribute__((ext_vector_type(8)))  float    v8f;

#define DEVFN static __device__ __forceinline__

// A-matrix fragment (16x32 f16), ISA 7.12.2 layout:
//   half = lane>>4, m = lane&15
//   elem j<8  -> K = k0 + 8*half + j
//   elem j>=8 -> K = k0 + 16 + 8*half + (j-8)
// -> two contiguous 16-byte loads per lane.
DEVFN v16h frag_a(const _Float16* base, int stride, int k0, int lane) {
  const int half = (lane >> 4) & 1;
  const int m    = lane & 15;
  const _Float16* p = base + (size_t)m * stride + k0 + half * 8;
  v8h lo = *(const v8h*)(p);
  v8h hi = *(const v8h*)(p + 16);
  v16h r;
#pragma unroll
  for (int i = 0; i < 8; ++i) { r[i] = lo[i]; r[i + 8] = hi[i]; }
  return r;
}

// B-matrix fragment (32x16 f16): lane col n = lane&15, half = lane>>4,
// elem j<8 -> K = k0 + 8*half + j, j>=8 -> K = k0 + 16 + 8*half + (j-8)
// (weights stored pre-transposed Wt[n*KP + k]).
DEVFN v16h frag_b(const _Float16* base, int stride, int k0, int lane) {
  const int half = (lane >> 4) & 1;
  const int n    = lane & 15;
  const _Float16* p = base + (size_t)n * stride + k0 + half * 8;
  v8h lo = *(const v8h*)(p);
  v8h hi = *(const v8h*)(p + 16);
  v16h r;
#pragma unroll
  for (int i = 0; i < 8; ++i) { r[i] = lo[i]; r[i + 8] = hi[i]; }
  return r;
}

// 32xKPx16 register-blocked tile: one B fragment is reused by two A tiles.
template <int KP>
DEVFN void tile_gemm2(const _Float16* A, int strideA, const _Float16* B,
                      int lane, v8f& c0, v8f& c1) {
  c0 = {};
  c1 = {};
#pragma unroll
  for (int k0 = 0; k0 < KP; k0 += 32) {
    v16h b  = frag_b(B, KP, k0, lane);
    v16h a0 = frag_a(A, strideA, k0, lane);
    v16h a1 = frag_a(A + (size_t)16 * strideA, strideA, k0, lane);
    c0 = __builtin_amdgcn_wmma_f32_16x16x32_f16(false, a0, false, b, (short)0,
                                                c0, false, false);
    c1 = __builtin_amdgcn_wmma_f32_16x16x32_f16(false, a1, false, b, (short)0,
                                                c1, false, false);
  }
}

// C/D layout: lane -> N = lane&15, VGPR v -> M = v + 8*(lane>>4).
// bias + leaky_relu(0.01) + f16 store into LDS activation tile.
DEVFN void store_act(_Float16* out, int strideO, int n0, int lane, v8f c,
                     const float* bias) {
  const int n    = n0 + (lane & 15);
  const int half = (lane >> 4) & 1;
  const float bb = bias[n];
#pragma unroll
  for (int v = 0; v < 8; ++v) {
    float x = c[v] + bb;
    x = x > 0.f ? x : 0.01f * x;
    out[(size_t)(v + half * 8) * strideO + n] = (_Float16)x;
  }
}

// MODE 0: node encoder   in = embed_w[x_type]            (K=7 ->32)  out_a = node (write)
// MODE 1: edge encoder   in = edge_attr                  (K=21->32)  out_a = edge (write)
// MODE 2: message        in = [node[dst],node[src],edge] (K=384)     out_a = edge (+= m),
//                                                                    out_b = aggr (atomic += m*dist)
// MODE 3: node update    in = [node, aggr]               (K=256)     out_a = node (+=)
// MODE 4: decoder        in = node                       (K=128)     out_a = d_out (N x 2)
template <int MODE>
__global__ __launch_bounds__(256) void mlp3_kernel(
    const int* x_type, const float* embed_w, const float* edge_attr,
    const int* e_src, const int* e_dst,
    const float* node_in, const float* edge_in, const float* aggr_in,
    const float* node_dist,
    const _Float16* __restrict__ W1, const float* b1,
    const _Float16* __restrict__ W2, const float* b2,
    const _Float16* __restrict__ W3, const float* b3,
    float* out_a, float* out_b, int nrows) {
  constexpr int K1  = (MODE == 2) ? 384 : (MODE == 3) ? 256 : (MODE == 4) ? 128 : 32;
  constexpr int K1R = (MODE == 0) ? 7 : (MODE == 1) ? 21 : K1;  // real gather width
  constexpr int S1  = K1 + 8;   // LDS stride pad: +16B keeps alignment, shifts banks
  constexpr int SH  = 136;
  constexpr int N3R = (MODE == 4) ? 2 : 128;
  constexpr int ROWS = 32;

  __shared__ __align__(16) _Float16 sA[ROWS * S1];
  __shared__ __align__(16) _Float16 sH1[ROWS * SH];
  __shared__ __align__(16) _Float16 sH2[ROWS * SH];

  const int tid  = threadIdx.x;
  const int wave = tid >> 5;
  const int lane = tid & 31;
  const int r0   = blockIdx.x * ROWS;

  // ---- stage 32 input rows into LDS (gather + f32->f16) ----
  for (int idx = tid; idx < ROWS * K1; idx += 256) {
    const int r = idx / K1, c = idx % K1;
    const int row = r0 + r;
    float v = 0.f;
    if (row < nrows) {
      if (MODE == 0) {
        if (c < K1R) v = embed_w[x_type[row] * 7 + c];
      } else if (MODE == 1) {
        if (c < K1R) v = edge_attr[(size_t)row * 21 + c];
      } else if (MODE == 2) {
        if (c < 128)      v = node_in[(size_t)e_dst[row] * 128 + c];
        else if (c < 256) v = node_in[(size_t)e_src[row] * 128 + (c - 128)];
        else              v = edge_in[(size_t)row * 128 + (c - 256)];
      } else if (MODE == 3) {
        v = (c < 128) ? node_in[(size_t)row * 128 + c]
                      : aggr_in[(size_t)row * 128 + (c - 128)];
      } else {
        v = node_in[(size_t)row * 128 + c];
      }
    }
    sA[(size_t)r * S1 + c] = (_Float16)v;
  }
  __syncthreads();

  const int n0 = wave * 16;

  // ---- layer 1 ----
  {
    v8f c0, c1;
    tile_gemm2<K1>(sA, S1, W1 + (size_t)n0 * K1, lane, c0, c1);
    store_act(sH1, SH, n0, lane, c0, b1);
    store_act(sH1 + (size_t)16 * SH, SH, n0, lane, c1, b1);
  }
  __syncthreads();
  // ---- layer 2 ----
  {
    v8f c0, c1;
    tile_gemm2<128>(sH1, SH, W2 + (size_t)n0 * 128, lane, c0, c1);
    store_act(sH2, SH, n0, lane, c0, b2);
    store_act(sH2 + (size_t)16 * SH, SH, n0, lane, c1, b2);
  }
  __syncthreads();
  // ---- layer 3 + mode epilogue ----
  if (!(MODE == 4 && wave > 0)) {
    v8f c0, c1;
    tile_gemm2<128>(sH2, SH, W3 + (size_t)n0 * 128, lane, c0, c1);
    const int n    = n0 + (lane & 15);
    const int half = (lane >> 4) & 1;
    const float bb = (n < N3R) ? b3[n] : 0.f;
#pragma unroll
    for (int t = 0; t < 2; ++t) {
      const v8f& c = t ? c1 : c0;
#pragma unroll
      for (int v = 0; v < 8; ++v) {
        const int row = r0 + t * 16 + v + 8 * half;
        if (row >= nrows) continue;
        const float x = c[v] + bb;
        if (MODE == 0 || MODE == 1) {
          out_a[(size_t)row * 128 + n] = x;
        } else if (MODE == 2) {
          out_a[(size_t)row * 128 + n] += x;  // edge += m (rows exclusive to block)
          atomicAdd(&out_b[(size_t)e_dst[row] * 128 + n], x * node_dist[row]);
        } else if (MODE == 3) {
          out_a[(size_t)row * 128 + n] += x;  // node += delta
        } else {
          if (n < 2) out_a[(size_t)row * 2 + n] = x;
        }
      }
    }
  }
}

// W (K x N, f32 row-major) -> Wt (NP x KP, f16, transposed + zero-padded)
__global__ __launch_bounds__(256) void convert_weight_kernel(
    const float* __restrict__ W, _Float16* __restrict__ Wt, int K, int N,
    int KP, int total) {
  const int idx = blockIdx.x * 256 + threadIdx.x;
  if (idx >= total) return;
  const int n = idx / KP, k = idx % KP;
  const float v = (k < K && n < N) ? W[(size_t)k * N + n] : 0.f;
  Wt[idx] = (_Float16)v;
}

extern "C" void kernel_launch(void* const* d_in, const int* in_sizes, int n_in,
                              void* d_out, int out_size, void* d_ws,
                              size_t ws_size, hipStream_t stream) {
  const int Nn = in_sizes[0];
  const int Ee = in_sizes[1] / 2;
  const int L = 10, H = 128;

  const int*   x_type    = (const int*)d_in[0];
  const int*   e_src     = (const int*)d_in[1];        // edge_index[0] = source
  const int*   e_dst     = e_src + Ee;                 // edge_index[1] = target
  const float* edge_attr = (const float*)d_in[2];
  const float* node_dist = (const float*)d_in[3];
  const float* embed_w   = (const float*)d_in[4];
  const float* P[30];
  for (int i = 0; i < 30; ++i) P[i] = (const float*)d_in[5 + i];
  // P[0..5]=ni W1,b1,W2,b2,W3,b3  P[6..11]=ei  P[12..17]=no
  // P[18..23]=me (stacked L)      P[24..29]=mn (stacked L)

  // ---- workspace layout: f16 weights, then f32 node/aggr/edge buffers ----
  _Float16* wp = (_Float16*)d_ws;
  size_t off = 0;
  auto take = [&](size_t nelem) { _Float16* p = wp + off; off += nelem; return p; };
  _Float16* niW1t = take(128 * 32);
  _Float16* niW2t = take(128 * 128);
  _Float16* niW3t = take(128 * 128);
  _Float16* eiW1t = take(128 * 32);
  _Float16* eiW2t = take(128 * 128);
  _Float16* eiW3t = take(128 * 128);
  _Float16* noW1t = take(128 * 128);
  _Float16* noW2t = take(128 * 128);
  _Float16* noW3t = take(16 * 128);
  _Float16* meW1t = take((size_t)L * 128 * 384);
  _Float16* meW2t = take((size_t)L * 128 * 128);
  _Float16* meW3t = take((size_t)L * 128 * 128);
  _Float16* mnW1t = take((size_t)L * 128 * 256);
  _Float16* mnW2t = take((size_t)L * 128 * 128);
  _Float16* mnW3t = take((size_t)L * 128 * 128);

  const size_t wbytes = (off * sizeof(_Float16) + 255) & ~(size_t)255;
  float* fnode = (float*)((char*)d_ws + wbytes);
  float* faggr = fnode + (size_t)Nn * H;
  float* fedge = faggr + (size_t)Nn * H;
  const size_t need = wbytes + ((size_t)2 * Nn * H + (size_t)Ee * H) * sizeof(float);
  if (need > ws_size) return;  // insufficient scratch (won't happen at these sizes)

  auto cvt = [&](const float* W, _Float16* Wt, int K, int Nc, int KP, int NP) {
    const int total = KP * NP;
    convert_weight_kernel<<<(total + 255) / 256, 256, 0, stream>>>(W, Wt, K, Nc,
                                                                   KP, total);
  };
  cvt(P[0], niW1t, 7, 128, 32, 128);
  cvt(P[2], niW2t, 128, 128, 128, 128);
  cvt(P[4], niW3t, 128, 128, 128, 128);
  cvt(P[6], eiW1t, 21, 128, 32, 128);
  cvt(P[8], eiW2t, 128, 128, 128, 128);
  cvt(P[10], eiW3t, 128, 128, 128, 128);
  cvt(P[12], noW1t, 128, 128, 128, 128);
  cvt(P[14], noW2t, 128, 128, 128, 128);
  cvt(P[16], noW3t, 128, 2, 128, 16);
  for (int l = 0; l < L; ++l) {
    cvt(P[18] + (size_t)l * 384 * 128, meW1t + (size_t)l * 128 * 384, 384, 128, 384, 128);
    cvt(P[20] + (size_t)l * 128 * 128, meW2t + (size_t)l * 128 * 128, 128, 128, 128, 128);
    cvt(P[22] + (size_t)l * 128 * 128, meW3t + (size_t)l * 128 * 128, 128, 128, 128, 128);
    cvt(P[24] + (size_t)l * 256 * 128, mnW1t + (size_t)l * 128 * 256, 256, 128, 256, 128);
    cvt(P[26] + (size_t)l * 128 * 128, mnW2t + (size_t)l * 128 * 128, 128, 128, 128, 128);
    cvt(P[28] + (size_t)l * 128 * 128, mnW3t + (size_t)l * 128 * 128, 128, 128, 128, 128);
  }

  const dim3 blk(256);
  const int gN = (Nn + 31) / 32;
  const int gE = (Ee + 31) / 32;

  // encoders
  mlp3_kernel<0><<<gN, blk, 0, stream>>>(
      x_type, embed_w, nullptr, nullptr, nullptr, nullptr, nullptr, nullptr,
      nullptr, niW1t, P[1], niW2t, P[3], niW3t, P[5], fnode, nullptr, Nn);
  mlp3_kernel<1><<<gE, blk, 0, stream>>>(
      nullptr, nullptr, edge_attr, nullptr, nullptr, nullptr, nullptr, nullptr,
      nullptr, eiW1t, P[7], eiW2t, P[9], eiW3t, P[11], fedge, nullptr, Ee);

  // L message-passing layers
  for (int l = 0; l < L; ++l) {
    hipMemsetAsync(faggr, 0, (size_t)Nn * H * sizeof(float), stream);
    mlp3_kernel<2><<<gE, blk, 0, stream>>>(
        nullptr, nullptr, nullptr, e_src, e_dst, fnode, fedge, nullptr,
        node_dist, meW1t + (size_t)l * 128 * 384, P[19] + (size_t)l * H,
        meW2t + (size_t)l * 128 * 128, P[21] + (size_t)l * H,
        meW3t + (size_t)l * 128 * 128, P[23] + (size_t)l * H, fedge, faggr, Ee);
    mlp3_kernel<3><<<gN, blk, 0, stream>>>(
        nullptr, nullptr, nullptr, nullptr, nullptr, fnode, nullptr, faggr,
        nullptr, mnW1t + (size_t)l * 128 * 256, P[25] + (size_t)l * H,
        mnW2t + (size_t)l * 128 * 128, P[27] + (size_t)l * H,
        mnW3t + (size_t)l * 128 * 128, P[29] + (size_t)l * H, fnode, nullptr, Nn);
  }

  // decoder -> d_out (N x 2, f32)
  mlp3_kernel<4><<<gN, blk, 0, stream>>>(
      nullptr, nullptr, nullptr, nullptr, nullptr, fnode, nullptr, nullptr,
      nullptr, noW1t, P[13], noW2t, P[15], noW3t, P[17], (float*)d_out, nullptr,
      Nn);
}
